// RingStarFuse_36017595744542
// MI455X (gfx1250) — matrix operand order repaired and checked by reading, not compiled, not run on hardware
//
#include <hip/hip_runtime.h>
#include <math.h>

// ---------------------------------------------------------------------------
// CDNA5 (gfx1250) implementation of RingStarFuse.
// Heavy linear algebra -> bf16 WMMA (v_wmma_f32_16x16x32_bf16), f32 accum.
// Global->LDS staging uses async-to-LDS (ASYNCcnt); K-chunk 64 => 2 WMMA per
// barrier round (halves sync overhead per FLOP vs K-chunk 32).
// ---------------------------------------------------------------------------

typedef __attribute__((ext_vector_type(16))) __bf16 v16bf;
typedef __attribute__((ext_vector_type(8)))  float  v8f;
typedef __attribute__((ext_vector_type(4)))  int    i32x4;

union FragBF { v16bf v; uint4 q[2]; };

#define DEV static __device__ __forceinline__

DEV int imin(int a, int b) { return a < b ? a : b; }
DEV int imax(int a, int b) { return a > b ? a : b; }

DEV unsigned short f2bf(float f) {               // f32 -> bf16 RNE
  unsigned u = __float_as_uint(f);
  unsigned r = u + 0x7FFFu + ((u >> 16) & 1u);
  return (unsigned short)(r >> 16);
}
DEV float siluf_(float x) { return x / (1.f + __expf(-x)); }

// ---- async global -> LDS (CDNA5) with safe fallback -----------------------
#if defined(__has_builtin)
#if __has_builtin(__builtin_amdgcn_global_load_async_to_lds_b128)
#define HAVE_ASYNC_LDS 1
#endif
#endif

DEV void copy_b128_g2l(const unsigned short* g, unsigned short* l) {
#ifdef HAVE_ASYNC_LDS
  __builtin_amdgcn_global_load_async_to_lds_b128(
      (__attribute__((address_space(1))) i32x4*)(i32x4*)(void*)g,
      (__attribute__((address_space(3))) i32x4*)(i32x4*)(void*)l, 0, 0);
#else
  *(uint4*)l = *(const uint4*)g;
#endif
}
DEV void wait_async_all() {
#ifdef HAVE_ASYNC_LDS
#if __has_builtin(__builtin_amdgcn_s_wait_asynccnt)
  __builtin_amdgcn_s_wait_asynccnt(0);
#else
  asm volatile("s_wait_asynccnt 0" ::: "memory");
#endif
#endif
}

DEV float bilin_clamp(const float* img, int H, int W, float yy, float xx) {
  float y0f = floorf(yy), x0f = floorf(xx);
  float wy = yy - y0f, wx = xx - x0f;
  int y0 = imin(imax((int)y0f, 0), H - 1), y1 = imin(imax((int)y0f + 1, 0), H - 1);
  int x0 = imin(imax((int)x0f, 0), W - 1), x1 = imin(imax((int)x0f + 1, 0), W - 1);
  float a = img[y0 * W + x0], b = img[y0 * W + x1];
  float c = img[y1 * W + x0], d = img[y1 * W + x1];
  return (1.f - wy) * ((1.f - wx) * a + wx * b) + wy * ((1.f - wx) * c + wx * d);
}

#define PI_F 3.14159265358979323846f
#define TWO_PI_F 6.28318530717958647692f
#define LROW 72   /* LDS row stride in ushorts: 64 k-elems + 8 pad (144B) */

// ===========================================================================
// Packing kernels
// ===========================================================================
// x (64,1024,14,14) f32 -> padded NHWC bf16 (64,16,16,1024)
__global__ void k_pack_xpad(const float* __restrict__ x, unsigned short* __restrict__ xp) {
  int idx = blockIdx.x * blockDim.x + threadIdx.x;
  const int TOT = 64 * 16 * 16 * 1024;
  if (idx >= TOT) return;
  int ci = idx & 1023;
  int t = idx >> 10;
  int px = t & 15; t >>= 4;
  int py = t & 15; int b = t >> 4;
  float v = 0.f;
  if (py >= 1 && py <= 14 && px >= 1 && px <= 14)
    v = x[(((size_t)b * 1024 + ci) * 14 + (py - 1)) * 14 + (px - 1)];
  xp[idx] = f2bf(v);
}

// generic: w (N,K) row-major f32 -> Bt (Np,Kp) bf16 zero padded
__global__ void k_pack_bt(const float* __restrict__ w, unsigned short* __restrict__ bt,
                          int N, int K, int Kp, int tot) {
  int idx = blockIdx.x * blockDim.x + threadIdx.x;
  if (idx >= tot) return;
  int k = idx % Kp, n = idx / Kp;
  float v = (n < N && k < K) ? w[(size_t)n * K + k] : 0.f;
  bt[idx] = f2bf(v);
}

// conv1 weights (64,1024,3,3) -> Bt[n][k], k=(ky*3+kx)*1024+ci
__global__ void k_pack_bt_conv1(const float* __restrict__ w, unsigned short* __restrict__ bt) {
  int idx = blockIdx.x * blockDim.x + threadIdx.x;
  const int TOT = 64 * 9216;
  if (idx >= TOT) return;
  int k = idx % 9216, n = idx / 9216;
  int kk = k >> 10, ci = k & 1023, ky = kk / 3, kx = kk - ky * 3;
  bt[idx] = f2bf(w[(((size_t)n * 1024 + ci) * 3 + ky) * 3 + kx]);
}

__global__ void k_bnprep(const float* g, const float* bb, const float* rm, const float* rv,
                         const float* bconv, float* scale, float* shift) {
  int c = threadIdx.x;
  float s = g[c] * rsqrtf(rv[c] + 1e-5f);
  scale[c] = s;
  shift[c] = (bconv[c] - rm[c]) * s + bb[c];
}

// ===========================================================================
// conv1 (1024->64, 3x3, pad 1) + BN as implicit GEMM with WMMA bf16
// M=12544 N=64 K=9216, K-chunk 64 (2 WMMA / stage). A from NHWC-padded bf16
// => every K-chunk is one contiguous 128B run -> pure async b128 staging.
// ===========================================================================
__global__ void __launch_bounds__(128) k_conv1_wmma(
    const unsigned short* __restrict__ xpad,  // (64,16,16,1024) bf16
    const unsigned short* __restrict__ Bt,    // (64,9216) bf16
    const float* __restrict__ scale, const float* __restrict__ shift,
    float* __restrict__ out)                  // (64,64,14,14) f32
{
  __shared__ unsigned short As[2][16 * LROW];
  __shared__ unsigned short Bs[2][64 * LROW];
  __shared__ unsigned rb[16];
  const int tid = threadIdx.x, lane = tid & 31, wave = tid >> 5;
  const int mt = blockIdx.x << 4;
  if (tid < 16) {
    int p = mt + tid, b = p / 196, r = p - b * 196, oy = r / 14, ox = r - oy * 14;
    rb[tid] = ((unsigned)((b * 16 + oy) * 16 + ox)) << 10;
  }
  __syncthreads();

  auto stage = [&](int kt, int buf) {
    int kk = kt >> 4, ci0 = (kt & 15) << 6;      // 16 chunks of 64 per (ky,kx)
    int ky = kk / 3, kx = kk - ky * 3;
    unsigned koff = ((unsigned)(ky * 16 + kx)) << 10;
    {  // A: 16 rows x 128B = 128 b128, one per thread
      int m = tid >> 3, q = tid & 7;
      copy_b128_g2l(xpad + rb[m] + koff + ci0 + q * 8, &As[buf][m * LROW + q * 8]);
    }
    for (int c = tid; c < 512; c += 128) {  // B: 64 rows x 128B
      int r = c >> 3, q = c & 7;
      copy_b128_g2l(Bt + (size_t)r * 9216 + (kt << 6) + q * 8, &Bs[buf][r * LROW + q * 8]);
    }
  };

  v8f acc = {0.f, 0.f, 0.f, 0.f, 0.f, 0.f, 0.f, 0.f};
  const int NKT = 144;   // 9216 / 64
  stage(0, 0);
  wait_async_all();
  __syncthreads();
  for (int kt = 0; kt < NKT; ++kt) {
    int buf = kt & 1;
    if (kt + 1 < NKT) stage(kt + 1, buf ^ 1);
    int m = lane & 15, h2 = lane >> 4;
    int nn = (wave << 4) + m;
#pragma unroll
    for (int s = 0; s < 2; ++s) {
      FragBF fa, fb;
      const unsigned short* pa = &As[buf][m * LROW + s * 32 + h2 * 8];
      fa.q[0] = *(const uint4*)pa;
      fa.q[1] = *(const uint4*)(pa + 16);
      const unsigned short* pb = &Bs[buf][nn * LROW + s * 32 + h2 * 8];
      fb.q[0] = *(const uint4*)pb;
      fb.q[1] = *(const uint4*)(pb + 16);
      acc = __builtin_amdgcn_wmma_f32_16x16x32_bf16(false, fa.v, false, fb.v,
                                                    (short)0, acc, false, false);
    }
    wait_async_all();
    __syncthreads();
  }
  int col = (wave << 4) + (lane & 15), h2 = lane >> 4;
  float sc = scale[col], sh = shift[col];
#pragma unroll
  for (int v = 0; v < 8; ++v) {
    int p = mt + h2 * 8 + v;
    int b = p / 196, r = p - b * 196;
    out[((size_t)b * 64 + col) * 196 + r] = acc[v] * sc + sh;
  }
}

// ===========================================================================
// Generic GEMM: C[M,N] = A[M,K](f32,ldA) @ Bt[Np,Kp](bf16)^T + bias
// Kp must be a multiple of 64 (async B loads read full 64-wide chunks).
// ===========================================================================
__global__ void __launch_bounds__(128) k_gemm_wmma(
    const float* __restrict__ Am, int ldA,
    const unsigned short* __restrict__ Bt, int Kp,
    float* __restrict__ Cm, int ldC,
    const float* __restrict__ bias, int M, int N, int K)
{
  __shared__ unsigned short As[2][16 * LROW];
  __shared__ unsigned short Bs[2][64 * LROW];
  const int tid = threadIdx.x, lane = tid & 31, wave = tid >> 5;
  const int mt = blockIdx.x << 4, nt = blockIdx.y << 6;
  const int nKt = (K + 63) >> 6;

  auto stage = [&](int kt, int buf) {
    const int k0 = kt << 6;
    {  // A: 1024 bf16, 8 per thread, convert f32->bf16 in flight
      int i = tid << 3, m = i >> 6, k = i & 63, row = mt + m;
      float f[8];
#pragma unroll
      for (int j = 0; j < 8; ++j) f[j] = 0.f;
      if (row < M) {
        const float* a = Am + (size_t)row * ldA + k0 + k;
        __builtin_prefetch(a + 128, 0, 0);
#pragma unroll
        for (int j = 0; j < 8; ++j)
          if (k0 + k + j < K) f[j] = a[j];
      }
      ushort4 h0, h1;
      h0.x = f2bf(f[0]); h0.y = f2bf(f[1]); h0.z = f2bf(f[2]); h0.w = f2bf(f[3]);
      h1.x = f2bf(f[4]); h1.y = f2bf(f[5]); h1.z = f2bf(f[6]); h1.w = f2bf(f[7]);
      *(ushort4*)&As[buf][m * LROW + k] = h0;
      *(ushort4*)&As[buf][m * LROW + k + 4] = h1;
    }
    for (int c = tid; c < 512; c += 128) {  // B: 64 rows x 128B, async b128
      int r = c >> 3, q = c & 7;
      copy_b128_g2l(Bt + (size_t)(nt + r) * Kp + k0 + q * 8, &Bs[buf][r * LROW + q * 8]);
    }
  };

  v8f acc = {0.f, 0.f, 0.f, 0.f, 0.f, 0.f, 0.f, 0.f};
  stage(0, 0);
  wait_async_all();
  __syncthreads();
  for (int kt = 0; kt < nKt; ++kt) {
    int buf = kt & 1;
    if (kt + 1 < nKt) stage(kt + 1, buf ^ 1);
    int m = lane & 15, h2 = lane >> 4;
    int nn = (wave << 4) + m;
#pragma unroll
    for (int s = 0; s < 2; ++s) {
      FragBF fa, fb;
      const unsigned short* pa = &As[buf][m * LROW + s * 32 + h2 * 8];
      fa.q[0] = *(const uint4*)pa;
      fa.q[1] = *(const uint4*)(pa + 16);
      const unsigned short* pb = &Bs[buf][nn * LROW + s * 32 + h2 * 8];
      fb.q[0] = *(const uint4*)pb;
      fb.q[1] = *(const uint4*)(pb + 16);
      acc = __builtin_amdgcn_wmma_f32_16x16x32_bf16(false, fa.v, false, fb.v,
                                                    (short)0, acc, false, false);
    }
    wait_async_all();
    __syncthreads();
  }
  int nl = lane & 15, h2 = lane >> 4, col = nt + (wave << 4) + nl;
  if (col < N) {
    float bv = bias ? bias[col] : 0.f;
#pragma unroll
    for (int v = 0; v < 8; ++v) {
      int row = mt + h2 * 8 + v;
      if (row < M) Cm[(size_t)row * ldC + col] = acc[v] + bv;
    }
  }
}

// ===========================================================================
// Ring attention pieces
// ===========================================================================
// tokens[(b*64+c), d] = bilinear polar sample of last rw columns + pos[d]
__global__ void k_ring_tokens(const float* __restrict__ x0, const float* __restrict__ pos,
                              float* __restrict__ tok, int Hp, int Wp, int rw, int D, int tot) {
  int idx = blockIdx.x * blockDim.x + threadIdx.x;
  if (idx >= tot) return;
  int d = idx % D;
  int t = idx / D;
  int c = t & 63, b = t >> 6;
  int ty = d / rw, j = d - ty * rw;
  int col = Wp - rw + j;
  float th = TWO_PI_F * (float)ty / (float)Hp;
  float r = (float)col + 0.5f;  // radius == Wp for all rings
  float yy = 6.5f + r * __sinf(th), xx = 6.5f + r * __cosf(th);
  const float* img = x0 + ((size_t)b * 64 + c) * 196;
  tok[idx] = bilin_clamp(img, 14, 14, yy, xx) + pos[d];
}

// single-head attention over 64 channel-tokens; qkv (4096, 3D) -> out (4096, D)
__global__ void k_attn(const float* __restrict__ qkv, float* __restrict__ o, int D) {
  int b = blockIdx.x, i = threadIdx.x;
  const float* base = qkv + (size_t)b * 64 * 3 * D;
  const float* q = base + (size_t)i * 3 * D;
  float lg[64];
  float scl = rsqrtf((float)D);
  for (int j = 0; j < 64; ++j) {
    const float* kk = base + (size_t)j * 3 * D + D;
    float s = 0.f;
    for (int d = 0; d < D; ++d) s += q[d] * kk[d];
    lg[j] = s * scl;
  }
  float mx = lg[0];
  for (int j = 1; j < 64; ++j) mx = fmaxf(mx, lg[j]);
  float se = 0.f;
  for (int j = 0; j < 64; ++j) { lg[j] = __expf(lg[j] - mx); se += lg[j]; }
  float inv = 1.f / se;
  float* op = o + ((size_t)b * 64 + i) * D;
  for (int d = 0; d < D; ++d) {
    float s = 0.f;
    for (int j = 0; j < 64; ++j) s += lg[j] * base[((size_t)j * 3 + 2) * D + d];
    op[d] = s * inv;
  }
}

// polar (b,C,Hp,Wp) -> cartesian (b,C,14,14); mask r > radius
__global__ void k_polar2cart(const float* __restrict__ p, float* __restrict__ out,
                             int C, int Hp, int Wp, float radius, int tot) {
  int idx = blockIdx.x * blockDim.x + threadIdx.x;
  if (idx >= tot) return;
  int x = idx % 14;
  int t = idx / 14;
  int y = t % 14; t /= 14;
  int c = t % C, b = t / C;
  float dy = (float)y - 6.5f, dx = (float)x - 6.5f;
  float r = sqrtf(dx * dx + dy * dy);
  float th = atan2f(dy, dx);
  if (th < 0.f) th += TWO_PI_F;
  float ti = th / TWO_PI_F * (float)Hp;
  float ri = r / radius * (float)Wp - 0.5f;
  float t0 = floorf(ti), wt = ti - t0;
  int t0i = ((int)t0) % Hp; if (t0i < 0) t0i += Hp;
  int t1i = (t0i + 1) % Hp;
  float r0 = floorf(ri), wr = ri - r0;
  int r0i = imin(imax((int)r0, 0), Wp - 1), r1i = imin(imax((int)r0 + 1, 0), Wp - 1);
  const float* pc = p + ((size_t)b * C + c) * Hp * Wp;
  float v = (1.f - wt) * ((1.f - wr) * pc[t0i * Wp + r0i] + wr * pc[t0i * Wp + r1i]) +
            wt * ((1.f - wr) * pc[t1i * Wp + r0i] + wr * pc[t1i * Wp + r1i]);
  out[idx] = (r <= radius) ? v : 0.f;
}

__global__ void k_mul(float* a, const float* b, int tot) {
  int i = blockIdx.x * blockDim.x + threadIdx.x;
  if (i < tot) a[i] *= b[i];
}
__global__ void k_add(float* a, const float* b, int tot) {
  int i = blockIdx.x * blockDim.x + threadIdx.x;
  if (i < tot) a[i] += b[i];
}

// ===========================================================================
// Small direct convolutions
// ===========================================================================
__global__ void k_conv_direct(const float* __restrict__ in, const float* __restrict__ w,
                              const float* __restrict__ bias, float* __restrict__ out,
                              int Cin, int Cout, int Kh, int pad, int groups,
                              int coff, int ctot, int tot) {
  int idx = blockIdx.x * blockDim.x + threadIdx.x;
  if (idx >= tot) return;
  int ox = idx % 14;
  int t = idx / 14;
  int oy = t % 14; t /= 14;
  int oc = t % Cout, b = t / Cout;
  int cpg = Cin / groups, opg = Cout / groups, g = oc / opg;
  float s = bias[oc];
  for (int ci = 0; ci < cpg; ++ci) {
    const float* ip = in + ((size_t)b * Cin + g * cpg + ci) * 196;
    const float* wp = w + ((size_t)oc * cpg + ci) * Kh * Kh;
    for (int ky = 0; ky < Kh; ++ky) {
      int iy = oy + ky - pad;
      if (iy < 0 || iy > 13) continue;
      for (int kx = 0; kx < Kh; ++kx) {
        int ix = ox + kx - pad;
        if (ix < 0 || ix > 13) continue;
        s += ip[iy * 14 + ix] * wp[ky * Kh + kx];
      }
    }
  }
  out[((size_t)b * ctot + coff + oc) * 196 + oy * 14 + ox] = s;
}

// depthwise 3x3 + residual x0
__global__ void k_dw2_res(const float* __restrict__ zc, const float* __restrict__ w,
                          const float* __restrict__ bias, const float* __restrict__ x0,
                          float* __restrict__ out) {
  int idx = blockIdx.x * blockDim.x + threadIdx.x;
  if (idx >= 64 * 64 * 196) return;
  int ox = idx % 14;
  int t = idx / 14;
  int oy = t % 14; t /= 14;
  int c = t % 64, b = t / 64;
  float s = bias[c];
  const float* ip = zc + ((size_t)b * 64 + c) * 196;
  for (int ky = 0; ky < 3; ++ky) {
    int iy = oy + ky - 1;
    if (iy < 0 || iy > 13) continue;
    for (int kx = 0; kx < 3; ++kx) {
      int ix = ox + kx - 1;
      if (ix < 0 || ix > 13) continue;
      s += ip[iy * 14 + ix] * w[c * 9 + ky * 3 + kx];
    }
  }
  out[idx] = x0[idx] + s;
}

// ===========================================================================
// circle_mamba helpers
// ===========================================================================
__global__ void k_cart2polar(const float* __restrict__ in, float* __restrict__ out,
                             int C, int Hp, int Wp, float radius, int tot) {
  int idx = blockIdx.x * blockDim.x + threadIdx.x;
  if (idx >= tot) return;
  int x = idx % Wp;
  int t = idx / Wp;
  int y = t % Hp; t /= Hp;
  int c = t % C, b = t / C;
  float th = TWO_PI_F * (float)y / (float)Hp;
  float r = ((float)x + 0.5f) * radius / (float)Wp;
  float yy = 6.5f + r * __sinf(th), xx = 6.5f + r * __cosf(th);
  out[idx] = bilin_clamp(in + ((size_t)b * C + c) * 196, 14, 14, yy, xx);
}

__global__ void k_resize(const float* __restrict__ in, float* __restrict__ out,
                         int Hi, int Wi, int Ho, int Wo, int tot) {
  int idx = blockIdx.x * blockDim.x + threadIdx.x;
  if (idx >= tot) return;
  int x = idx % Wo;
  int t = idx / Wo;
  int y = t % Ho;
  int bc = t / Ho;
  float sy = ((float)y + 0.5f) * (float)Hi / (float)Ho - 0.5f;
  float sx = ((float)x + 0.5f) * (float)Wi / (float)Wo - 0.5f;
  sy = fminf(fmaxf(sy, 0.f), (float)(Hi - 1));
  sx = fminf(fmaxf(sx, 0.f), (float)(Wi - 1));
  out[idx] = bilin_clamp(in + (size_t)bc * Hi * Wi, Hi, Wi, sy, sx);
}

__global__ void k_rotate(const float* __restrict__ in, float* __restrict__ out,
                         int H, int W, float ct, float st, int tot) {
  int idx = blockIdx.x * blockDim.x + threadIdx.x;
  if (idx >= tot) return;
  int x = idx % W;
  int t = idx / W;
  int y = t % H;
  int bc = t / H;
  float cy = (H - 1) * 0.5f, cx = (W - 1) * 0.5f;
  float dy = (float)y - cy, dx = (float)x - cx;
  float sx = ct * dx + st * dy + cx;
  float sy = -st * dx + ct * dy + cy;
  float sxr = rintf(sx), syr = rintf(sy);
  int xi = imin(imax((int)sxr, 0), W - 1);
  int yi = imin(imax((int)syr, 0), H - 1);
  float ok = (sxr >= 0.f && sxr <= (float)(W - 1) && syr >= 0.f && syr <= (float)(H - 1)) ? 1.f : 0.f;
  out[idx] = in[(size_t)bc * H * W + yi * W + xi] * ok;
}

// im2col for 7x7/stride7 patch embed: A[(b*9+t)][ci*49+py*7+px]
__global__ void k_pe_im2col(const float* __restrict__ vimg, float* __restrict__ A, int tot) {
  int idx = blockIdx.x * blockDim.x + threadIdx.x;
  if (idx >= tot) return;
  int k = idx % 3136, m = idx / 3136;
  int ci = k / 49, rem = k - ci * 49, py = rem / 7, px = rem - py * 7;
  int b = m / 9, t9 = m - b * 9, ty = t9 / 3, tx = t9 - ty * 3;
  A[idx] = vimg[(((size_t)b * 64 + ci) * 22 + ty * 7 + py) * 22 + tx * 7 + px];
}

__global__ void k_build_tok(const float* __restrict__ tpe, const float* __restrict__ cls,
                            const float* __restrict__ pos, float* __restrict__ tok, int tot) {
  int idx = blockIdx.x * blockDim.x + threadIdx.x;
  if (idx >= tot) return;
  int d = idx % 384;
  int t = idx / 384;
  int l = t % 10, b = t / 10;
  float v = (l == 0) ? cls[d] : tpe[((size_t)b * 9 + l - 1) * 384 + d];
  tok[idx] = v + pos[l * 384 + d];
}

__global__ void k_rmsnorm(const float* __restrict__ in, const float* __restrict__ w,
                          float* __restrict__ out, int rows, int d) {
  int r = blockIdx.x * blockDim.x + threadIdx.x;
  if (r >= rows) return;
  const float* x = in + (size_t)r * d;
  float ss = 0.f;
  for (int i = 0; i < d; ++i) ss += x[i] * x[i];
  float inv = rsqrtf(ss / (float)d + 1e-5f);
  float* o = out + (size_t)r * d;
  for (int i = 0; i < d; ++i) o[i] = x[i] * inv * w[i];
}

// causal depthwise conv1d (k=4) + SiLU; xi = xz[:, :768]; flip for backward dir
__global__ void k_conv1d_silu(const float* __restrict__ xz, const float* __restrict__ cw,
                              const float* __restrict__ cb, float* __restrict__ u,
                              int flip, int tot) {
  int idx = blockIdx.x * blockDim.x + threadIdx.x;
  if (idx >= tot) return;
  int di = idx % 768;
  int t = idx / 768;
  int l = t % 10, b = t / 10;
  float s = cb[di];
#pragma unroll
  for (int q = 0; q < 4; ++q) {
    int ls = l - 3 + q;
    if (ls >= 0) {
      int lp = flip ? 9 - ls : ls;
      s += xz[((size_t)(b * 10 + lp)) * 1536 + di] * cw[di * 4 + q];
    }
  }
  u[idx] = siluf_(s);
}

__global__ void k_softplus(float* p, int tot) {
  int i = blockIdx.x * blockDim.x + threadIdx.x;
  if (i < tot) {
    float x = p[i];
    p[i] = (x > 20.f) ? x : log1pf(__expf(x));
  }
}

// selective scan: n=16 states, L=10, per (b, di) thread
__global__ void k_ssm_scan(const float* __restrict__ dtb, const float* __restrict__ u,
                           const float* __restrict__ proj, const float* __restrict__ A_log,
                           const float* __restrict__ Dp, float* __restrict__ ys, int tot) {
  int idx = blockIdx.x * blockDim.x + threadIdx.x;
  if (idx >= tot) return;
  int di = idx % 768, b = idx / 768;
  float An[16], h[16];
#pragma unroll
  for (int n = 0; n < 16; ++n) { An[n] = -__expf(A_log[di * 16 + n]); h[n] = 0.f; }
  float Dv = Dp[di];
  for (int l = 0; l < 10; ++l) {
    size_t r = (size_t)(b * 10 + l);
    float dt = dtb[r * 768 + di];
    float uv = u[r * 768 + di];
    const float* pr = proj + r * 56;
    float y = 0.f;
#pragma unroll
    for (int n = 0; n < 16; ++n) {
      float dA = __expf(dt * An[n]);
      h[n] = dA * h[n] + dt * pr[24 + n] * uv;
      y += h[n] * pr[40 + n];
    }
    ys[r * 768 + di] = y + uv * Dv;
  }
}

__global__ void k_combine(const float* __restrict__ ysf, const float* __restrict__ ysb,
                          const float* __restrict__ xz, float* __restrict__ yc, int tot) {
  int idx = blockIdx.x * blockDim.x + threadIdx.x;
  if (idx >= tot) return;
  int di = idx % 768;
  int t = idx / 768;
  int l = t % 10, b = t / 10;
  float yf = ysf[idx];
  float yb = ysb[((size_t)(b * 10 + (9 - l))) * 768 + di];
  float zg = xz[((size_t)(b * 10 + l)) * 1536 + 768 + di];
  yc[idx] = (yf + yb) * siluf_(zg);
}

__global__ void k_mean_tok(const float* __restrict__ fn, float* __restrict__ feat, int tot) {
  int idx = blockIdx.x * blockDim.x + threadIdx.x;
  if (idx >= tot) return;
  int d = idx % 384, b = idx / 384;
  float s = 0.f;
  for (int l = 0; l < 10; ++l) s += fn[((size_t)(b * 10 + l)) * 384 + d];
  feat[idx] = s * 0.1f;
}

// out = clip(2*z2 + pol_broadcast, 0, 6)
__global__ void k_final(const float* __restrict__ z2, const float* __restrict__ pol,
                        float* __restrict__ out, int tot) {
  int idx = blockIdx.x * blockDim.x + threadIdx.x;
  if (idx >= tot) return;
  int hw = idx % 196;
  int b = idx / (196 * 64);
  float v = 2.f * z2[idx] + pol[(size_t)b * 196 + hw];
  out[idx] = fminf(fmaxf(v, 0.f), 6.f);
}

// ===========================================================================
// Host orchestration
// ===========================================================================
static inline int G(long n) { return (int)((n + 255) / 256); }
static inline int up(int v, int m) { return (v + m - 1) / m * m; }

static void gemm(hipStream_t s, const float* A, int ldA, const unsigned short* Bt, int Kp,
                 float* C, int ldC, const float* bias, int M, int N, int K) {
  dim3 g((M + 15) / 16, (N + 63) / 64);
  k_gemm_wmma<<<g, 128, 0, s>>>(A, ldA, Bt, Kp, C, ldC, bias, M, N, K);
}
static void packbt(hipStream_t s, const float* w, unsigned short* bt, int N, int K,
                   int Np, int Kp) {
  int tot = Np * Kp;
  k_pack_bt<<<G(tot), 256, 0, s>>>(w, bt, N, K, Kp, tot);
}

// input index map (depth-first, dict insertion order of setup_inputs)
#define IN_X 0
#define IN_DW 1      /* w,b,bn_g,bn_b,bn_rm,bn_rv */
#define IN_RA1 7     /* pos,wqkv,bqkv,wo,bo */
#define IN_CP 22     /* w1,b1,w2,b2,w3,b3,w4,b4 */
#define IN_DW2 30    /* w,b */
#define IN_PEW 32
#define IN_PEB 33
#define IN_CLS 34
#define IN_POS 35
#define IN_BLK0 36   /* 8 blocks x 17 tensors */
#define IN_NORMF 172
#define IN_HEADW 173
#define IN_HEADB 174

extern "C" void kernel_launch(void* const* d_in, const int* in_sizes, int n_in,
                              void* d_out, int out_size, void* d_ws, size_t ws_size,
                              hipStream_t stream) {
  (void)in_sizes; (void)n_in; (void)out_size; (void)ws_size;
#define FP(i) ((const float*)d_in[(i)])

  // ---- workspace bump allocator (~150 MB) ----
  char* base = (char*)d_ws;
  size_t off = 0;
  auto AF = [&](size_t n) -> float* {
    float* p = (float*)(base + off);
    off = (off + n * 4 + 255) & ~(size_t)255;
    return p;
  };
  auto AH = [&](size_t n) -> unsigned short* {
    unsigned short* p = (unsigned short*)(base + off);
    off = (off + n * 2 + 255) & ~(size_t)255;
    return p;
  };

  unsigned short* XPAD = AH((size_t)64 * 16 * 16 * 1024);
  unsigned short* BT1  = AH((size_t)64 * 9216);
  unsigned short* BTQ  = AH((size_t)1408 * 512);
  unsigned short* BTO  = AH((size_t)512 * 512);
  unsigned short* BTPE = AH((size_t)384 * 3136);
  unsigned short* BTIN = AH((size_t)1536 * 384);
  unsigned short* BTXP = AH((size_t)64 * 768);
  unsigned short* BTDT = AH((size_t)768 * 64);
  unsigned short* BTOW = AH((size_t)384 * 768);
  unsigned short* BTHD = AH((size_t)512 * 384);

  float* SCALE = AF(64); float* SHIFT = AF(64);
  float* X0  = AF(802816);
  float* TOKR = AF((size_t)4096 * 456);
  float* QKV = AF((size_t)4096 * 1368);
  float* ATT = AF((size_t)4096 * 456);
  float* PRJ = AF((size_t)4096 * 456);
  float* XA = AF(802816); float* XB = AF(802816);
  float* ZC = AF(802816); float* Z2 = AF(802816);
  float* VP1 = AF((size_t)64 * 64 * 22 * 22);
  float* VP2 = AF((size_t)64 * 64 * 22 * 22);
  float* APE = AF((size_t)576 * 3136);
  float* TPE = AF((size_t)576 * 384);
  float* TOK = AF((size_t)640 * 384);
  float* XN  = AF((size_t)640 * 384);
  float* XZ  = AF((size_t)640 * 1536);
  float* UD  = AF((size_t)640 * 768);
  float* PS  = AF((size_t)640 * 56);
  float* DTB = AF((size_t)640 * 768);
  float* YSF = AF((size_t)640 * 768);
  float* YSB = AF((size_t)640 * 768);
  float* YC  = AF((size_t)640 * 768);
  float* DLT = AF((size_t)640 * 384);
  float* FEAT = AF((size_t)64 * 384);
  float* YIMG = AF((size_t)64 * 484);
  float* YR1 = AF((size_t)64 * 22 * 7);
  float* YR2 = AF((size_t)64 * 22 * 7);
  float* POL = AF((size_t)64 * 196);

  // ---- stage 1: entry conv 1024->64 + BN via WMMA implicit GEMM ----
  k_pack_xpad<<<G((long)64 * 16 * 16 * 1024), 256, 0, stream>>>(FP(IN_X), XPAD);
  k_pack_bt_conv1<<<G((long)64 * 9216), 256, 0, stream>>>(FP(IN_DW + 0), BT1);
  k_bnprep<<<1, 64, 0, stream>>>(FP(IN_DW + 2), FP(IN_DW + 3), FP(IN_DW + 4),
                                 FP(IN_DW + 5), FP(IN_DW + 1), SCALE, SHIFT);
  k_conv1_wmma<<<784, 128, 0, stream>>>(XPAD, BT1, SCALE, SHIFT, X0);

  // ---- stage 2: three ring attentions, product into XA ----
  const int raHp[3] = {22, 31, 38}, raWp[3] = {7, 10, 12}, raRw[3] = {2, 2, 12};
  for (int i = 0; i < 3; ++i) {
    int Hp = raHp[i], Wp = raWp[i], rw = raRw[i], D = Hp * rw;
    int ib = IN_RA1 + 5 * i;
    int Kp = up(D, 64);
    packbt(stream, FP(ib + 1), BTQ, 3 * D, D, up(3 * D, 64), Kp);
    packbt(stream, FP(ib + 3), BTO, D, D, up(D, 64), Kp);
    int tot = 4096 * D;
    k_ring_tokens<<<G(tot), 256, 0, stream>>>(X0, FP(ib + 0), TOKR, Hp, Wp, rw, D, tot);
    gemm(stream, TOKR, D, BTQ, Kp, QKV, 3 * D, FP(ib + 2), 4096, 3 * D, D);
    k_attn<<<64, 64, 0, stream>>>(QKV, ATT, D);
    gemm(stream, ATT, D, BTO, Kp, PRJ, D, FP(ib + 4), 4096, D, D);
    float* dst = (i == 0) ? XA : XB;
    k_polar2cart<<<G(802816), 256, 0, stream>>>(PRJ, dst, 64, D, 1, (float)Wp, 802816);
    if (i) k_mul<<<G(802816), 256, 0, stream>>>(XA, XB, 802816);
  }

  // ---- stage 3: multi-kernel conv branches + depthwise + residual ----
  k_conv_direct<<<G((long)64 * 16 * 196), 256, 0, stream>>>(XA, FP(IN_CP + 0), FP(IN_CP + 1), ZC, 64, 16, 3, 1, 1, 0, 64, 64 * 16 * 196);
  k_conv_direct<<<G((long)64 * 16 * 196), 256, 0, stream>>>(XA, FP(IN_CP + 2), FP(IN_CP + 3), ZC, 64, 16, 5, 2, 4, 16, 64, 64 * 16 * 196);
  k_conv_direct<<<G((long)64 * 16 * 196), 256, 0, stream>>>(XA, FP(IN_CP + 4), FP(IN_CP + 5), ZC, 64, 16, 7, 3, 8, 32, 64, 64 * 16 * 196);
  k_conv_direct<<<G((long)64 * 16 * 196), 256, 0, stream>>>(XA, FP(IN_CP + 6), FP(IN_CP + 7), ZC, 64, 16, 9, 4, 16, 48, 64, 64 * 16 * 196);
  k_dw2_res<<<G(802816), 256, 0, stream>>>(ZC, FP(IN_DW2 + 0), FP(IN_DW2 + 1), X0, Z2);

  // ---- stage 4: circle_mamba ----
  k_cart2polar<<<G((long)64 * 64 * 22 * 7), 256, 0, stream>>>(Z2, VP2, 64, 22, 7, 7.f, 64 * 64 * 22 * 7);
  k_resize<<<G((long)64 * 64 * 22 * 22), 256, 0, stream>>>(VP2, VP1, 22, 7, 22, 22, 64 * 64 * 22 * 22);
  k_rotate<<<G((long)64 * 64 * 22 * 22), 256, 0, stream>>>(VP1, VP2, 22, 22, 0.f, -1.f, 64 * 64 * 22 * 22);
  k_pe_im2col<<<G((long)576 * 3136), 256, 0, stream>>>(VP2, APE, 576 * 3136);
  packbt(stream, FP(IN_PEW), BTPE, 384, 3136, 384, 3136);
  gemm(stream, APE, 3136, BTPE, 3136, TPE, 384, FP(IN_PEB), 576, 384, 3136);
  k_build_tok<<<G((long)640 * 384), 256, 0, stream>>>(TPE, FP(IN_CLS), FP(IN_POS), TOK, 640 * 384);

  for (int k = 0; k < 8; ++k) {
    int ib = IN_BLK0 + k * 17;
    k_rmsnorm<<<(640 + 127) / 128, 128, 0, stream>>>(TOK, FP(ib + 0), XN, 640, 384);
    packbt(stream, FP(ib + 1), BTIN, 1536, 384, 1536, 384);
    gemm(stream, XN, 384, BTIN, 384, XZ, 1536, nullptr, 640, 1536, 384);
    for (int dir = 0; dir < 2; ++dir) {
      int id = ib + 3 + dir * 7;  // conv_w,conv_b,xproj_w,dt_w,dt_b,A_log,D
      k_conv1d_silu<<<G((long)640 * 768), 256, 0, stream>>>(XZ, FP(id + 0), FP(id + 1), UD, dir, 640 * 768);
      packbt(stream, FP(id + 2), BTXP, 56, 768, 64, 768);
      gemm(stream, UD, 768, BTXP, 768, PS, 56, nullptr, 640, 56, 768);
      packbt(stream, FP(id + 3), BTDT, 768, 24, 768, 64);
      gemm(stream, PS, 56, BTDT, 64, DTB, 768, FP(id + 4), 640, 768, 24);
      k_softplus<<<G((long)640 * 768), 256, 0, stream>>>(DTB, 640 * 768);
      k_ssm_scan<<<G((long)64 * 768), 256, 0, stream>>>(DTB, UD, PS, FP(id + 5), FP(id + 6),
                                                        dir ? YSB : YSF, 64 * 768);
    }
    k_combine<<<G((long)640 * 768), 256, 0, stream>>>(YSF, YSB, XZ, YC, 640 * 768);
    packbt(stream, FP(ib + 2), BTOW, 384, 768, 384, 768);
    gemm(stream, YC, 768, BTOW, 768, DLT, 384, nullptr, 640, 384, 768);
    k_add<<<G((long)640 * 384), 256, 0, stream>>>(TOK, DLT, 640 * 384);
  }

  k_rmsnorm<<<(640 + 127) / 128, 128, 0, stream>>>(TOK, FP(IN_NORMF), XN, 640, 384);
  k_mean_tok<<<G((long)64 * 384), 256, 0, stream>>>(XN, FEAT, 64 * 384);
  packbt(stream, FP(IN_HEADW), BTHD, 484, 384, 512, 384);
  gemm(stream, FEAT, 384, BTHD, 384, YIMG, 484, FP(IN_HEADB), 64, 484, 384);

  k_resize<<<G((long)64 * 22 * 7), 256, 0, stream>>>(YIMG, YR1, 22, 22, 22, 7, 64 * 22 * 7);
  k_rotate<<<G((long)64 * 22 * 7), 256, 0, stream>>>(YR1, YR2, 22, 7, 0.f, 1.f, 64 * 22 * 7);
  k_polar2cart<<<G((long)64 * 196), 256, 0, stream>>>(YR2, POL, 1, 22, 7, 7.f, 64 * 196);

  // final: z = zb + circle_mamba(zb) = 2*Z2 + pol ; relu6
  k_final<<<G(802816), 256, 0, stream>>>(Z2, POL, (float*)d_out, 802816);
#undef FP
}
